// DoubleExtraTrain_53223234732120
// MI455X (gfx1250) — compile-verified, matrix operand order checked by reading, not run on hardware
//
#include <hip/hip_runtime.h>
#include <hip/hip_bf16.h>

#define B_TOT 2048
#define T_LEN 128
#define F_IN  32
#define U_N   48
#define HST   64    // padded f16 state stride (tail 48..63 stays zero)
#define S_N   15
#define TU    144   // 3*U
#define DD    24

typedef __attribute__((ext_vector_type(16))) _Float16 v16h;
typedef __attribute__((ext_vector_type(8)))  float    v8f;

__device__ __forceinline__ v8f vsplat8(float v) {
    v8f z = {v, v, v, v, v, v, v, v};
    return z;
}

__device__ __forceinline__ v8f wmma16x16x32(v16h a, v16h b, v8f c) {
    // D = A(16x32 f16) * B(32x16 f16) + C(16x16 f32)
    return __builtin_amdgcn_wmma_f32_16x16x32_f16(false, a, false, b,
                                                  (short)0, c, false, false);
}

// K-index for element i of a 16-bit 16x32 A/B fragment (ISA 7.12.2):
// VGPR j=i/2, pair p=i%2; lanes 0-15 hold K {0..7,16..23}, lanes 16-31 {8..15,24..31}
__device__ __forceinline__ int frag_k(int i, int lane) {
    int j = i >> 1, p = i & 1;
    return ((j >> 2) << 4) + (((lane >> 4) & 1) << 3) + ((j & 3) << 1) + p;
}

// A fragment (16 x 32 tile at k0, rows = lane&15) from padded f16 LDS.
__device__ __forceinline__ v16h a_frag_h(const _Float16* src, int ld, int k0, int lane) {
    v16h a; int m = lane & 15;
#pragma unroll
    for (int i = 0; i < 16; ++i) a[i] = src[m * ld + k0 + frag_k(i, lane)];
    return a;
}

// B fragment (32 x 16 tile at column n0) from row-major f32 global weights,
// zero-padded past kmax. One-time cost before the time loop.
__device__ __forceinline__ v16h b_frag_f32(const float* w, int ld, int k0, int n0, int kmax, int lane) {
    v16h b; int n = lane & 15;
#pragma unroll
    for (int i = 0; i < 16; ++i) {
        int k = k0 + frag_k(i, lane);
        b[i] = (k < kmax) ? (_Float16)w[k * ld + n0 + n] : (_Float16)0.0f;
    }
    return b;
}

// Inverse map (row m, column col) -> index in fragment-layout buffer [tile][lane][8]
__device__ __forceinline__ int fragidx(int m, int col) {
    return ((col >> 4) << 8) + ((((m >> 3) << 4) | (col & 15)) << 3) + (m & 7);
}

// fast gate activations: single v_rcp_f32 instead of IEEE division sequence
__device__ __forceinline__ float sigf(float v) {
    return __builtin_amdgcn_rcpf(1.0f + __expf(-v));
}
__device__ __forceinline__ float tanh_fast(float v) {
    float r = __builtin_amdgcn_rcpf(1.0f + __expf(-2.0f * v));
    return __builtin_fmaf(2.0f, r, -1.0f);
}

__global__ __launch_bounds__(192, 1)
void gru_shared_fused(const float* __restrict__ x,
                      const float* __restrict__ card_k,  const float* __restrict__ card_rk,
                      const float* __restrict__ card_bi, const float* __restrict__ card_br,
                      const float* __restrict__ cat_k,   const float* __restrict__ cat_rk,
                      const float* __restrict__ cat_b,
                      const float* __restrict__ w1, const float* __restrict__ b1,
                      const float* __restrict__ w2, const float* __restrict__ b2,
                      float* __restrict__ out)
{
    __shared__ _Float16 s_x[16 * F_IN];                        // current x_t tile (f16)
    __shared__ int      s_ids[16];                             // category ids
    __shared__ __align__(16) _Float16 s_hcard[16 * HST];       // branch-1 hidden (f16, padded)
    __shared__ __align__(16) _Float16 s_hprev[16 * HST];       // gathered per-category state
    __shared__ __align__(16) _Float16 s_rh   [16 * HST];       // r * h_prev
    __shared__ __align__(16) _Float16 s_table[16 * S_N * HST]; // shared-state table (f16)
    __shared__ float    s_hcat [16 * U_N];                     // latest branch-2 output
    // fragment-native layouts: [tile][lane][8] -> vector ds stores/loads
    __shared__ float    s_gx  [9 * 256];                       // branch-1 x@k + bi
    __shared__ float    s_gh  [9 * 256];                       // branch-1 h@rk + br
    __shared__ float    s_z2f [3 * 256];                       // branch-2 z gate
    __shared__ float    s_r2f [3 * 256];                       // branch-2 r gate
    __shared__ float    s_xh2f[3 * 256];                       // branch-2 x@k_h + b_h

    const int  tid  = threadIdx.x;
    const int  lane = tid & 31;
    const int  wv   = tid >> 5;     // 0..5 (wave32)
    const int  wsub = wv % 3;       // tile group within branch
    const bool br2  = (wv >= 3);
    const int  r0   = blockIdx.x * 16;

    // ---- register-resident weight fragments (loaded once, f32 -> f16) ----
    const float* Kw  = br2 ? cat_k  : card_k;
    const float* RKw = br2 ? cat_rk : card_rk;
    v16h BK[3], BRK0[3], BRK1[3];
    float bx[3], bh[3];
#pragma unroll
    for (int tt = 0; tt < 3; ++tt) {
        int n0 = (3 * wsub + tt) * 16;
        BK[tt]   = b_frag_f32(Kw,  TU, 0,  n0, F_IN, lane);
        BRK0[tt] = b_frag_f32(RKw, TU, 0,  n0, U_N,  lane);  // K 0..31
        BRK1[tt] = b_frag_f32(RKw, TU, 32, n0, U_N,  lane);  // K 32..47 (zero-padded)
        int col = n0 + (lane & 15);
        bx[tt] = br2 ? cat_b[col] : card_bi[col];
        bh[tt] = br2 ? 0.0f       : card_br[col];
    }

    // ---- zero-init recurrent state (incl. padded tails, never rewritten) ----
    for (int e = tid; e < 16 * HST; e += 192) {
        s_hcard[e] = (_Float16)0.0f; s_hprev[e] = (_Float16)0.0f; s_rh[e] = (_Float16)0.0f;
    }
    for (int e = tid; e < 16 * U_N; e += 192) s_hcat[e] = 0.f;
    for (int e = tid; e < 16 * S_N * HST; e += 192) s_table[e] = (_Float16)0.0f;
    __syncthreads();

    for (int t = 0; t < T_LEN; ++t) {
        // ---- stage 0: stage x_t tile to LDS (f16), ids, prefetch next step ----
        for (int e = tid; e < 16 * F_IN; e += 192) {
            int m = e >> 5, f = e & 31;
            s_x[e] = (_Float16)x[((size_t)(r0 + m) * T_LEN + t) * F_IN + f];
        }
        if (tid < 16) {
            int id = (int)x[((size_t)(r0 + tid) * T_LEN + t) * F_IN + 2];
            s_ids[tid] = id < 0 ? 0 : (id > S_N - 1 ? S_N - 1 : id);
            if (t + 1 < T_LEN)
                __builtin_prefetch(&x[((size_t)(r0 + tid) * T_LEN + (t + 1)) * F_IN], 0, 3);
        }
        __syncthreads();

        // ---- stage 0b: gather per-category state, one 128-bit LDS copy per lane ----
        if (br2) {
            int l = tid - 96;           // 0..95 = 16 rows x 6 segments of 8 f16
            int m = l / 6, seg = l % 6;
            const uint4* srcp = (const uint4*)&s_table[(m * S_N + s_ids[m]) * HST];
            ((uint4*)&s_hprev[m * HST])[seg] = srcp[seg];
        }
        __syncthreads();

        // ---- phase 1: WMMA matmuls (wave-uniform control => EXEC all-ones) ----
        {
            v16h ax = a_frag_h(s_x, F_IN, 0, lane);
            const int fb = lane << 3;               // fragment base within a tile
            if (!br2) {
                // branch 1: mx = x@k + bi ; mh = h@rk + br  (bias folded into C init)
                v16h ah0 = a_frag_h(s_hcard, HST, 0,  lane);
                v16h ah1 = a_frag_h(s_hcard, HST, 32, lane);
#pragma unroll
                for (int tt = 0; tt < 3; ++tt) {
                    int g = 3 * wsub + tt;          // global column tile 0..8
                    v8f cx = vsplat8(bx[tt]);
                    cx = wmma16x16x32(ax, BK[tt], cx);
                    v8f ch = vsplat8(bh[tt]);
                    ch = wmma16x16x32(ah0, BRK0[tt], ch);
                    ch = wmma16x16x32(ah1, BRK1[tt], ch);
#pragma unroll
                    for (int r = 0; r < 8; ++r) {
                        s_gx[g * 256 + fb + r] = cx[r];
                        s_gh[g * 256 + fb + r] = ch[r];
                    }
                }
            } else if (wv != 5) {
                // branch 2 gates: z (wave 3) / r (wave 4) = sigmoid(x@k + b + h@rk)
                v16h ah0 = a_frag_h(s_hprev, HST, 0,  lane);
                v16h ah1 = a_frag_h(s_hprev, HST, 32, lane);
                float* dst = (wv == 3) ? s_z2f : s_r2f;
#pragma unroll
                for (int tt = 0; tt < 3; ++tt) {
                    v8f c = vsplat8(bx[tt]);
                    c = wmma16x16x32(ax,  BK[tt],   c);
                    c = wmma16x16x32(ah0, BRK0[tt], c);
                    c = wmma16x16x32(ah1, BRK1[tt], c);
#pragma unroll
                    for (int r = 0; r < 8; ++r)
                        dst[tt * 256 + fb + r] = sigf(c[r]);
                }
            } else {
                // wave 5: candidate input part xh = x@k_h + b_h
#pragma unroll
                for (int tt = 0; tt < 3; ++tt) {
                    v8f c = vsplat8(bx[tt]);
                    c = wmma16x16x32(ax, BK[tt], c);
#pragma unroll
                    for (int r = 0; r < 8; ++r)
                        s_xh2f[tt * 256 + fb + r] = c[r];
                }
            }
        }
        __syncthreads();

        // ---- phase 2: branch-1 elementwise update ; branch-2 r*h ----
        if (!br2) {
            for (int e = tid; e < 16 * U_N; e += 96) {
                int m = e / U_N, u = e % U_N;
                int i0 = fragidx(m, u), i1 = fragidx(m, 48 + u), i2 = fragidx(m, 96 + u);
                float z  = sigf(s_gx[i0] + s_gh[i0]);
                float r  = sigf(s_gx[i1] + s_gh[i1]);
                float hc = tanh_fast(s_gx[i2] + r * s_gh[i2]);
                float h  = (float)s_hcard[m * HST + u];
                s_hcard[m * HST + u] = (_Float16)(z * h + (1.f - z) * hc);
            }
        } else {
            for (int e = tid - 96; e < 16 * U_N; e += 96) {
                int m = e / U_N, u = e % U_N;
                s_rh[m * HST + u] =
                    (_Float16)(s_r2f[fragidx(m, u)] * (float)s_hprev[m * HST + u]);
            }
        }
        __syncthreads();

        // ---- phase 3: wave 5 — (r*h)@rk_h via WMMA, update, scatter to table ----
        if (wv == 5) {
            v16h arh0 = a_frag_h(s_rh, HST, 0,  lane);
            v16h arh1 = a_frag_h(s_rh, HST, 32, lane);
            const int n = lane & 15, mb = (lane >> 4) << 3, fb = lane << 3;
#pragma unroll
            for (int tt = 0; tt < 3; ++tt) {
                v8f c = vsplat8(0.0f);               // no bias on the recurrent h-part
                c = wmma16x16x32(arh0, BRK0[tt], c);
                c = wmma16x16x32(arh1, BRK1[tt], c);
                int ub = tt * 16 + n;
#pragma unroll
                for (int r = 0; r < 8; ++r) {
                    int m = mb + r;
                    // z and xh live at identical fragment coordinates
                    float hc = tanh_fast(s_xh2f[tt * 256 + fb + r] + c[r]);
                    float z  = s_z2f[tt * 256 + fb + r];
                    float hp = (float)s_hprev[m * HST + ub];
                    float hn = z * hp + (1.f - z) * hc;
                    s_hcat[m * U_N + ub] = hn;
                    s_table[(m * S_N + s_ids[m]) * HST + ub] = (_Float16)hn;
                }
            }
        }
        __syncthreads();
    }

    // ---- dense head: relu(concat(h_card,h_cat) @ w1 + b1) @ w2 + b2 -> sigmoid ----
    if (tid < 16) {
        int m = tid;
        float acc2 = b2[0];
        for (int j = 0; j < DD; ++j) {
            float a = b1[j];
            for (int i = 0; i < U_N; ++i) a += (float)s_hcard[m * HST + i] * w1[i * DD + j];
            for (int i = 0; i < U_N; ++i) a += s_hcat[m * U_N + i] * w1[(U_N + i) * DD + j];
            a = fmaxf(a, 0.f);
            acc2 += a * w2[j];
        }
        out[r0 + m] = sigf(acc2);
    }
}

extern "C" void kernel_launch(void* const* d_in, const int* in_sizes, int n_in,
                              void* d_out, int out_size, void* d_ws, size_t ws_size,
                              hipStream_t stream) {
    (void)in_sizes; (void)n_in; (void)out_size; (void)d_ws; (void)ws_size;
    const float* x       = (const float*)d_in[0];
    const float* card_k  = (const float*)d_in[1];
    const float* card_rk = (const float*)d_in[2];
    const float* card_bi = (const float*)d_in[3];
    const float* card_br = (const float*)d_in[4];
    const float* cat_k   = (const float*)d_in[5];
    const float* cat_rk  = (const float*)d_in[6];
    const float* cat_b   = (const float*)d_in[7];
    const float* w1      = (const float*)d_in[8];
    const float* b1      = (const float*)d_in[9];
    const float* w2      = (const float*)d_in[10];
    const float* b2      = (const float*)d_in[11];
    float* out = (float*)d_out;

    dim3 grid(B_TOT / 16), block(192);
    hipLaunchKernelGGL(gru_shared_fused, grid, block, 0, stream,
                       x, card_k, card_rk, card_bi, card_br,
                       cat_k, cat_rk, cat_b, w1, b1, w2, b2, out);
}
